// RNNJANSEN_34866544509351
// MI455X (gfx1250) — compile-verified, matrix operand order checked by reading, not run on hardware
//
#include <hip/hip_runtime.h>
#include <stdint.h>

#define N_ 512
#define T_ 20
#define S_ 300
#define BUF_ 400
#define O_ 64
#define RING_ 128          // delay <= 66 always (dist<100, conduct >= 1.5)
#define RPAD_ 129          // bank-conflict padding (stride 129 floats)
#define DT_ 0.0001f

typedef __attribute__((ext_vector_type(16))) _Float16 v16h;
typedef __attribute__((ext_vector_type(8)))  float    v8f;

// ---------- helpers ----------
__device__ __forceinline__ float relu_(float x) { return fmaxf(x, 0.f); }

__device__ __forceinline__ float h2f_bits(unsigned short u) {
    union { _Float16 h; unsigned short u; } cv; cv.u = u; return (float)cv.h;
}
__device__ __forceinline__ unsigned short f2h_bits(float x) {
    union { _Float16 h; unsigned short u; } cv; cv.h = (_Float16)x; return cv.u;
}
__device__ __forceinline__ float waveRed(float v) {
    #pragma unroll
    for (int o = 16; o; o >>= 1) v += __shfl_xor(v, o, 32);
    return v;
}

// ---------- prep kernels ----------
__global__ void kZero(float* sumsq) {
    if (threadIdx.x == 0 && blockIdx.x == 0) sumsq[0] = 0.f;
}

// w_l[i,j] = log1p(0.5*(exp(w_bb[i,j])*sc[i,j] + exp(w_bb[j,i])*sc[j,i])); accumulate Frobenius^2
__global__ void kWl(const float* __restrict__ w_bb, const float* __restrict__ sc,
                    float* __restrict__ wl, float* __restrict__ sumsq) {
    __shared__ float red[256];
    int i = blockIdx.x, tid = threadIdx.x;
    float local = 0.f;
    for (int j = tid; j < N_; j += 256) {
        float v = 0.5f * (expf(w_bb[i * N_ + j]) * sc[i * N_ + j] +
                          expf(w_bb[j * N_ + i]) * sc[j * N_ + i]);
        float w = log1pf(v);
        wl[i * N_ + j] = w;
        local += w * w;
    }
    red[tid] = local; __syncthreads();
    for (int o = 128; o; o >>= 1) { if (tid < o) red[tid] += red[tid + o]; __syncthreads(); }
    if (tid == 0) atomicAdd(sumsq, red[0]);
}

// packed[i,j] = (f16(w_n[i,j])<<16) | delay[j,i];  dgv[i] = rowsum(w_n)
__global__ void kPack(const float* __restrict__ wl, const float* __restrict__ dist,
                      const float* __restrict__ theta, uint32_t* __restrict__ packed,
                      float* __restrict__ dgv, const float* __restrict__ sumsq) {
    __shared__ float red[256];
    int i = blockIdx.x, tid = threadIdx.x;
    float inv = 1.0f / sqrtf(sumsq[0]);
    float cond = 1.5f + relu_(theta[16]);     // CONDUCT_LB + relu(mu)
    float rsum = 0.f;
    for (int j = tid; j < N_; j += 256) {
        float wn = wl[i * N_ + j] * inv;
        rsum += wn;
        int del = (int)(dist[j * N_ + i] / cond);   // delays[j,i]
        del = min(max(del, 0), BUF_ - 1);
        del = min(del, RING_ - 1);
        packed[i * N_ + j] = ((uint32_t)f2h_bits(wn) << 16) | (uint32_t)del;
    }
    red[tid] = rsum; __syncthreads();
    for (int o = 128; o; o >>= 1) { if (tid < o) red[tid] += red[tid + o]; __syncthreads(); }
    if (tid == 0) dgv[i] = red[0];
}

// rs[o] = sum_j |lm[o,j]|
__global__ void kLmRs(const float* __restrict__ lm, float* __restrict__ rs) {
    __shared__ float red[256];
    int o = blockIdx.x, tid = threadIdx.x;
    float local = 0.f;
    for (int j = tid; j < N_; j += 256) local += fabsf(lm[o * N_ + j]);
    red[tid] = local; __syncthreads();
    for (int s = 128; s; s >>= 1) { if (tid < s) red[tid] += red[tid + s]; __syncthreads(); }
    if (tid == 0) rs[o] = red[0];
}

// lm16[o,j] = f16( lm[o,j]/rs[o] - mean_o(lm[o,j]/rs[o]) )
__global__ void kLm16(const float* __restrict__ lm, const float* __restrict__ rs,
                      _Float16* __restrict__ lm16) {
    int j = blockIdx.x * 256 + threadIdx.x;
    if (j >= N_) return;
    float mean = 0.f;
    for (int o = 0; o < O_; ++o) mean += lm[o * N_ + j] / rs[o];
    mean *= (1.0f / O_);
    for (int o = 0; o < O_; ++o)
        lm16[o * N_ + j] = (_Float16)(lm[o * N_ + j] / rs[o] - mean);
}

// ---------- gather helper: 4 packed elements (one uint4) -> acc ----------
__device__ __forceinline__ void gather4(uint4 p, int jbase, int head,
                                        const float* ring, float& acc) {
    int s0 = (head + (int)(p.x & 0xFFFFu)) & (RING_ - 1);
    int s1 = (head + (int)(p.y & 0xFFFFu)) & (RING_ - 1);
    int s2 = (head + (int)(p.z & 0xFFFFu)) & (RING_ - 1);
    int s3 = (head + (int)(p.w & 0xFFFFu)) & (RING_ - 1);
    acc = fmaf(h2f_bits((unsigned short)(p.x >> 16)), ring[(jbase + 0) * RPAD_ + s0], acc);
    acc = fmaf(h2f_bits((unsigned short)(p.y >> 16)), ring[(jbase + 1) * RPAD_ + s1], acc);
    acc = fmaf(h2f_bits((unsigned short)(p.z >> 16)), ring[(jbase + 2) * RPAD_ + s2], acc);
    acc = fmaf(h2f_bits((unsigned short)(p.w >> 16)), ring[(jbase + 3) * RPAD_ + s3], acc);
}

// ---------- persistent simulation kernel (1 workgroup, 32 wave32s, state in LDS) ----------
__global__ void __launch_bounds__(1024, 1)
kMain(const float* __restrict__ theta, const float* __restrict__ hx,
      const float* __restrict__ hE0, const float* __restrict__ external,
      const float* __restrict__ noise, const uint32_t* __restrict__ packed,
      const float* __restrict__ dgv, const _Float16* __restrict__ lm16,
      float* __restrict__ out) {
    __shared__ float ring[N_ * RPAD_];   // 264192 B : hE[j][d] = ring[j*129 + (head+d)&127]
    __shared__ float sM[N_], sE[N_], sI[N_], sMv[N_], sEv[N_], sIv[N_];
    __shared__ float sLEd[N_];
    __shared__ float sX[N_];

    const int tid = threadIdx.x;
    const int wave = tid >> 5, lane = tid & 31;

    // theta
    const float A  = theta[0],  a  = theta[1],  B  = theta[2],  b  = theta[3];
    const float g  = theta[4],  gf = theta[5],  gb = theta[6];
    const float c1 = theta[7],  c2 = theta[8],  c3 = theta[9],  c4 = theta[10];
    const float std_in = theta[11], vmax = theta[12], v0 = theta[13], r = theta[14];
    const float y0 = theta[15], kk = theta[17], ki = theta[18], cy0 = theta[19];
    const float gC = 0.01f + relu_(g), gF = 0.01f + relu_(gf), gB = 0.01f + relu_(gb);
    const float kC = (0.5f + relu_(kk)) * ki;
    const float s0 = relu_(std_in), s12 = 150.f + relu_(std_in);
    const float Aa = A * a, a2 = a * a, Bb = B * b, b2 = b * b;

    // init ring from hE0 (only first RING_ columns are ever readable)
    for (int idx = tid; idx < N_ * RING_; idx += 1024) {
        int j = idx >> 7, d = idx & (RING_ - 1);
        ring[j * RPAD_ + d] = hE0[j * BUF_ + d];
    }
    for (int i = tid; i < N_; i += 1024) {
        sM[i]  = hx[i * 6 + 0]; sE[i]  = hx[i * 6 + 1]; sI[i]  = hx[i * 6 + 2];
        sMv[i] = hx[i * 6 + 3]; sEv[i] = hx[i * 6 + 4]; sIv[i] = hx[i * 6 + 5];
    }
    __syncthreads();

    int head = 0;
    for (int tr = 0; tr < T_; ++tr) {
        for (int s = 0; s < S_; ++s) {
            // ---- LEd[i] = sum_j w_n[i,j] * hE[j, delays[j,i]]
            // wave owns rows 16w..16w+15, processed in pairs for ILP;
            // lane owns 4 consecutive j per uint4 (b128 loads of the packed w/delay stream)
            #pragma unroll 2
            for (int rr = 0; rr < 8; ++rr) {
                const int i0 = (wave << 4) + 2 * rr;
                const int i1 = i0 + 1;
                const uint4* r0 = (const uint4*)(packed + i0 * N_);
                const uint4* r1 = (const uint4*)(packed + i1 * N_);
                float acc0 = 0.f, acc1 = 0.f;
                #pragma unroll
                for (int t = 0; t < 4; ++t) {
                    uint4 p0 = r0[t * 32 + lane];
                    uint4 p1 = r1[t * 32 + lane];
                    const int jb = t * 128 + lane * 4;
                    gather4(p0, jb, head, ring, acc0);
                    gather4(p1, jb, head, ring, acc1);
                }
                acc0 = waveRed(acc0);
                acc1 = waveRed(acc1);
                if (lane == 0) { sLEd[i0] = acc0; sLEd[i1] = acc1; }
            }
            __syncthreads();

            const int nh = (head + RING_ - 1) & (RING_ - 1);
            if (tid < N_) {
                const int i = tid;
                float M = sM[i], E = sE[i], I = sI[i];
                float Mv = sMv[i], Ev = sEv[i], Iv = sIv[i];
                float LEd = sLEd[i], dgvi = dgv[i];
                float u = external[i * (S_ * T_) + s * T_ + tr];
                int nb = ((tr * S_ + s) * 3) * N_ + i;
                float n0 = noise[nb], n1 = noise[nb + N_], n2 = noise[nb + 2 * N_];
                float EI = E - I;
                float sg0 = vmax / (1.f + __expf(r * (v0 - EI)));
                float sg1 = vmax / (1.f + __expf(r * (v0 - c1 * M)));
                float sg2 = vmax / (1.f + __expf(r * (v0 - c3 * M)));
                float rM = kC * u + s0 * n0 + gC * (LEd - dgvi * M) + sg0;
                float rE = s12 * n1 + gF * (LEd - dgvi * EI) + c2 * sg1;
                float rI = s12 * n2 + gB * (dgvi * EI - LEd) + c4 * sg2;
                float Mn = fmaf(DT_, Mv, M), En = fmaf(DT_, Ev, E), In = fmaf(DT_, Iv, I);
                float uM = 500.f * tanhf(rM * (1.f / 500.f));
                float uE = 500.f * tanhf(rE * (1.f / 500.f));
                float uI = 500.f * tanhf(rI * (1.f / 500.f));
                float Mvn = Mv + DT_ * (Aa * uM - 2.f * a * Mv - a2 * M);
                float Evn = Ev + DT_ * (Aa * uE - 2.f * a * Ev - a2 * E);
                float Ivn = Iv + DT_ * (Bb * uI - 2.f * b * Iv - b2 * I);
                sM[i] = Mn; sE[i] = En; sI[i] = In;
                sMv[i] = Mvn; sEv[i] = Evn; sIv[i] = Ivn;
                ring[i * RPAD_ + nh] = Mn;   // prepend Mn to delay buffer
            }
            head = nh;
            __syncthreads();

            // ---- EEG sample: eeg = cy0 * (lm_t @ (E-I)) - y0, via v_wmma_f32_16x16x32_f16
            if (s == S_ - 1) {
                if (tid < N_) sX[tid] = sE[tid] - sI[tid];
                __syncthreads();
                if (wave < 4) {
                    const int m = wave;              // 16-row block of the 64-row lm_t
                    const int mr = lane & 15, hi = lane >> 4, kb = hi * 8;
                    v8f acc = {};
                    for (int t = 0; t < 16; ++t) {   // K = 512 in chunks of 32
                        v16h av = {}, bv = {};
                        const _Float16* arow = lm16 + (m * 16 + mr) * N_ + t * 32;
                        #pragma unroll
                        for (int v = 0; v < 4; ++v) {
                            av[2 * v]     = arow[kb + 2 * v];
                            av[2 * v + 1] = arow[kb + 2 * v + 1];
                            av[2 * v + 8] = arow[16 + kb + 2 * v];
                            av[2 * v + 9] = arow[16 + kb + 2 * v + 1];
                        }
                        if (mr == 0) {               // only column 0 of B is nonzero
                            #pragma unroll
                            for (int v = 0; v < 4; ++v) {
                                bv[2 * v]     = (_Float16)sX[t * 32 + kb + 2 * v];
                                bv[2 * v + 1] = (_Float16)sX[t * 32 + kb + 2 * v + 1];
                                bv[2 * v + 8] = (_Float16)sX[t * 32 + 16 + kb + 2 * v];
                                bv[2 * v + 9] = (_Float16)sX[t * 32 + 16 + kb + 2 * v + 1];
                            }
                        }
                        acc = __builtin_amdgcn_wmma_f32_16x16x32_f16(
                            false, av, false, bv, (short)0, acc, false, false);
                    }
                    if (mr == 0) {                   // D column 0: lanes 0 (rows 0..7), 16 (rows 8..15)
                        #pragma unroll
                        for (int q = 0; q < 8; ++q) {
                            int o = m * 16 + hi * 8 + q;
                            out[o * T_ + tr] = cy0 * acc[q] - y0;
                        }
                    }
                }
                __syncthreads();
            }
        }
    }
}

// ---------- launcher ----------
extern "C" void kernel_launch(void* const* d_in, const int* in_sizes, int n_in,
                              void* d_out, int out_size, void* d_ws, size_t ws_size,
                              hipStream_t stream) {
    const float* theta    = (const float*)d_in[0];
    const float* lm       = (const float*)d_in[1];
    const float* w_bb     = (const float*)d_in[2];
    const float* sc       = (const float*)d_in[3];
    const float* dist     = (const float*)d_in[4];
    const float* hx       = (const float*)d_in[5];
    const float* hE0      = (const float*)d_in[6];
    const float* external = (const float*)d_in[7];
    const float* noise    = (const float*)d_in[8];
    float* out = (float*)d_out;

    char* ws = (char*)d_ws;
    float*     sumsq  = (float*)(ws + 0);
    float*     dgv    = (float*)(ws + 256);
    float*     rs     = (float*)(ws + 2304);
    _Float16*  lm16   = (_Float16*)(ws + 4096);                 // 64 KB
    uint32_t*  packed = (uint32_t*)(ws + 69632);                // 1 MB
    float*     wl     = (float*)(ws + 69632 + 1048576);         // 1 MB

    kZero <<<1, 64, 0, stream>>>(sumsq);
    kWl   <<<N_, 256, 0, stream>>>(w_bb, sc, wl, sumsq);
    kPack <<<N_, 256, 0, stream>>>(wl, dist, theta, packed, dgv, sumsq);
    kLmRs <<<O_, 256, 0, stream>>>(lm, rs);
    kLm16 <<<2, 256, 0, stream>>>(lm, rs, lm16);
    kMain <<<1, 1024, 0, stream>>>(theta, hx, hE0, external, noise, packed, dgv, lm16, out);
}
